// Encoder_51127290692114
// MI455X (gfx1250) — matrix operand order repaired and checked by reading, not compiled, not run on hardware
//
#include <hip/hip_runtime.h>
#include <cmath>

// GNN encoder for MI455X (gfx1250, wave32, WMMA).
// Dense math: v_wmma_f32_16x16x32_f16 (f16 in, f32 accumulate).
// Both A and B fragments are fed by ds_load_b128 (W staged transposed in LDS).
// Edge stage fuses edge-GEMM + gather + sigmoid*softplus + atomic scatter.

typedef __attribute__((ext_vector_type(16))) _Float16 v16h;
typedef __attribute__((ext_vector_type(8)))  _Float16 v8h;
typedef __attribute__((ext_vector_type(8)))  float    v8f;

#define HID    128
#define NNODES 50000
#define NEDGES 800000
#define FNODE  92
#define FEDGE  50

__device__ __forceinline__ v8f wmma_f16(v16h a, v16h b, v8f c) {
  // (neg_a, A, neg_b, B, c_mod, C, reuse_a, reuse_b)
  return __builtin_amdgcn_wmma_f32_16x16x32_f16(false, a, false, b, (short)0, c,
                                                false, false);
}

// A fragment: lane holds row m = lane&15; elements are two contiguous 8-half
// runs at K = ko + 8*(lane>>4) and +16 (ISA 7.12.2 16-bit A layout).
// base = &Alds[m*kstride + 8*(lane>>4)] ; both runs 16B-aligned.
__device__ __forceinline__ v16h ld_afrag(const _Float16* base) {
  v8h lo = *(const v8h*)(base);
  v8h hi = *(const v8h*)(base + 16);
  return __builtin_shufflevector(lo, hi, 0, 1, 2, 3, 4, 5, 6, 7,
                                 8, 9, 10, 11, 12, 13, 14, 15);
}

// B fragment from TRANSPOSED weight tile WldsT[n][k]: lane holds column
// n = nb + (lane&15); elements are two contiguous 8-half runs at
// K = ko + 8*(lane>>4) and +16 (same as A).
// base = &WldsT[n*kstride + 8*(lane>>4)].
__device__ __forceinline__ v16h ld_bfrag(const _Float16* base) {
  v8h lo = *(const v8h*)(base);
  v8h hi = *(const v8h*)(base + 16);
  return __builtin_shufflevector(lo, hi, 0, 1, 2, 3, 4, 5, 6, 7,
                                 8, 9, 10, 11, 12, 13, 14, 15);
}

// C = act(A[MxK] @ W[Kx128] + bias), C[m*ldc + coff + n].
// 256 threads = 8 waves; block covers 64 rows x 128 cols; wave w owns cols
// [16w,16w+16) and loops over 4 M-tiles with the weight tile LDS-resident.
__global__ __launch_bounds__(256) void gemm128(
    const float* __restrict__ A, int lda,
    const float* __restrict__ W,
    const float* __restrict__ bias,
    float* __restrict__ C, int ldc, int coff,
    int M, int K, int KP, int relu)
{
  __shared__ __align__(16) _Float16 WldsT[128 * 128];  // [n=128][k=KP]
  __shared__ __align__(16) _Float16 Alds[64 * 128];    // [64][KP]
  const int tid   = threadIdx.x;
  const int mbase = blockIdx.x * 64;

  for (int idx = tid; idx < 128 * KP; idx += 256) {
    int n = idx / KP, k = idx - n * KP;
    WldsT[idx] = (_Float16)((k < K) ? W[k * 128 + n] : 0.0f);
  }
  for (int idx = tid; idx < 64 * KP; idx += 256) {
    int r = idx / KP, k = idx - r * KP;
    int m = mbase + r;
    Alds[idx] = (_Float16)((m < M && k < K) ? A[(long long)m * lda + k] : 0.0f);
  }
  __syncthreads();

  const int wave = tid >> 5, lane = tid & 31;
  const int g = lane >> 4;
  const int nn = wave * 16 + (lane & 15);
  const float bv = bias ? bias[nn] : 0.0f;
  const _Float16* bcol = WldsT + nn * KP + 8 * g;

#pragma unroll
  for (int mt = 0; mt < 4; ++mt) {
    const _Float16* arow = Alds + (mt * 16 + (lane & 15)) * KP + 8 * g;
    v8f acc = {};
    for (int ko = 0; ko < KP; ko += 32) {
      v16h a = ld_afrag(arow + ko);
      v16h b = ld_bfrag(bcol + ko);
      acc = wmma_f16(a, b, acc);
    }
#pragma unroll
    for (int r = 0; r < 8; ++r) {
      int m = mbase + mt * 16 + r + 8 * g;
      if (m < M) {
        float v = acc[r] + bv;
        if (relu) v = fmaxf(v, 0.0f);
        C[(long long)m * ldc + coff + nn] = v;
      }
    }
  }
}

// Per 64-edge block: proj = edge_attr(64x50) @ [Wfe | Wse](50x256) via WMMA,
// pf/ps = proj + gathered node projections from P[node][512],
// msg = sigmoid(pf)*softplus(ps), atomic scatter into agg[dst][128].
__global__ __launch_bounds__(256) void edge_kernel(
    const float* __restrict__ P,          // [NNODES][512]: f_i | f_j | s_i | s_j
    const float* __restrict__ edge_attr,  // [NEDGES][FEDGE]
    const int*   __restrict__ eidx,       // [2][NEDGES]: row0 = src, row1 = dst
    const float* __restrict__ Wfe,        // [FEDGE][128]
    const float* __restrict__ Wse,        // [FEDGE][128]
    const float* __restrict__ bf,         // [128]
    const float* __restrict__ bs,         // [128]
    float* __restrict__ agg,              // [NNODES][128]
    int nE)
{
  __shared__ __align__(16) _Float16 WldsT[256 * 64];  // [n=256][k=64]; n<128: Wfe, else Wse
  __shared__ __align__(16) _Float16 Elds[64 * 64];    // [64 edges][k=64] (50 -> 64 pad)
  __shared__ int srcL[64], dstL[64];
  const int tid = threadIdx.x;
  const long long e0 = (long long)blockIdx.x * 64;

  for (int idx = tid; idx < 256 * 64; idx += 256) {
    int n = idx >> 6, k = idx & 63;
    float v = 0.0f;
    if (k < FEDGE) v = (n < 128) ? Wfe[k * 128 + n] : Wse[k * 128 + (n - 128)];
    WldsT[idx] = (_Float16)v;
  }
  for (int idx = tid; idx < 64 * 64; idx += 256) {
    int r = idx >> 6, k = idx & 63;
    long long e = e0 + r;
    Elds[idx] = (_Float16)((e < nE && k < FEDGE) ? edge_attr[e * FEDGE + k] : 0.0f);
  }
  if (tid < 64) {
    long long e = e0 + tid;
    srcL[tid] = (e < nE) ? eidx[e] : 0;
    dstL[tid] = (e < nE) ? eidx[(long long)nE + e] : 0;
  }
  __syncthreads();

  const int wave = tid >> 5, lane = tid & 31;
  const int g = lane >> 4;
  const int nn = wave * 16 + (lane & 15);
  const float bfv = bf[nn], bsv = bs[nn];
  const _Float16* bcolF = WldsT + nn * 64 + 8 * g;
  const _Float16* bcolS = WldsT + (nn + 128) * 64 + 8 * g;

#pragma unroll
  for (int mt = 0; mt < 4; ++mt) {
    const _Float16* arow = Elds + (mt * 16 + (lane & 15)) * 64 + 8 * g;
    v8f accF = {}, accS = {};
#pragma unroll
    for (int ko = 0; ko < 64; ko += 32) {
      v16h a = ld_afrag(arow + ko);
      accF = wmma_f16(a, ld_bfrag(bcolF + ko), accF);
      accS = wmma_f16(a, ld_bfrag(bcolS + ko), accS);
    }
#pragma unroll
    for (int r = 0; r < 8; ++r) {
      int m = mt * 16 + r + 8 * g;
      if (e0 + m < nE) {
        int d = dstL[m], s = srcL[m];
        const float* Pd = P + (long long)d * 512;
        const float* Ps = P + (long long)s * 512;
        float pf = accF[r] + Pd[nn]       + Ps[128 + nn] + bfv;
        float ps = accS[r] + Pd[256 + nn] + Ps[384 + nn] + bsv;
        float sig = 1.0f / (1.0f + __expf(-pf));
        float sp  = (ps > 20.0f) ? ps : log1pf(__expf(ps));
        atomicAdd(agg + (long long)d * HID + nn, sig * sp);
      }
    }
  }
}

extern "C" void kernel_launch(void* const* d_in, const int* in_sizes, int n_in,
                              void* d_out, int out_size, void* d_ws, size_t ws_size,
                              hipStream_t stream)
{
  const float* x    = (const float*)d_in[0];
  const float* ea   = (const float*)d_in[1];
  const float* Win  = (const float*)d_in[2];
  const float* bin  = (const float*)d_in[3];
  const float* Wf   = (const float*)d_in[4];
  const float* bfp  = (const float*)d_in[5];
  const float* Ws   = (const float*)d_in[6];
  const float* bsp  = (const float*)d_in[7];
  const float* Wm   = (const float*)d_in[8];
  const float* bmp  = (const float*)d_in[9];
  const int*   eidx = (const int*)d_in[10];

  float* h   = (float*)d_ws;                      // [NNODES][128]
  float* agg = h + (size_t)NNODES * HID;          // [NNODES][128]
  float* P   = agg + (size_t)NNODES * HID;        // [NNODES][512]

  const dim3 blk(256);
  const int gM = (NNODES + 63) / 64;
  const int gE = (NEDGES + 63) / 64;
  const int WROWS = 2 * HID + FEDGE;  // 306

  // h = relu(x @ W_in + b_in); K=92 padded to 96
  gemm128<<<gM, blk, 0, stream>>>(x, FNODE, Win, bin, h, HID, 0, NNODES, FNODE, 96, 1);

  for (int l = 0; l < 2; ++l) {
    const float* Wfl = Wf + (size_t)l * WROWS * HID;
    const float* Wsl = Ws + (size_t)l * WROWS * HID;
    // P = [h@Wf_i | h@Wf_j | h@Ws_i | h@Ws_j]
    gemm128<<<gM, blk, 0, stream>>>(h, HID, Wfl,             nullptr, P, 512, 0,   NNODES, HID, HID, 0);
    gemm128<<<gM, blk, 0, stream>>>(h, HID, Wfl + HID * HID, nullptr, P, 512, 128, NNODES, HID, HID, 0);
    gemm128<<<gM, blk, 0, stream>>>(h, HID, Wsl,             nullptr, P, 512, 256, NNODES, HID, HID, 0);
    gemm128<<<gM, blk, 0, stream>>>(h, HID, Wsl + HID * HID, nullptr, P, 512, 384, NNODES, HID, HID, 0);
    // agg = h (residual), then scatter-add messages
    (void)hipMemcpyAsync(agg, h, (size_t)NNODES * HID * sizeof(float),
                         hipMemcpyDeviceToDevice, stream);
    edge_kernel<<<gE, blk, 0, stream>>>(P, ea, eidx,
        Wfl + 2 * HID * HID, Wsl + 2 * HID * HID,
        bfp + l * HID, bsp + l * HID, agg, NEDGES);
    // h = relu((h+agg) @ Wm + bm); last layer writes straight to d_out
    float* outp = (l == 1) ? (float*)d_out : h;
    gemm128<<<gM, blk, 0, stream>>>(agg, HID, Wm + (size_t)l * HID * HID,
                                    bmp + l * HID, outp, HID, 0, NNODES, HID, HID, 1);
  }
}
